// DiffusionRoiHead_90975997264564
// MI455X (gfx1250) — matrix operand order repaired and checked by reading, not compiled, not run on hardware
//
#include <hip/hip_runtime.h>
#include <math.h>

// ---------------- WMMA types (CDNA5 / gfx1250, wave32) ----------------
typedef __attribute__((ext_vector_type(16))) __bf16 bf16x16;
typedef __attribute__((ext_vector_type(8)))  float  f32x8;

__device__ __forceinline__ __bf16 cvt_bf16(float x) { return (__bf16)x; }

__device__ __forceinline__ void pack8(bf16x16& f, int base, float4 a, float4 b)
{
  f[base + 0] = cvt_bf16(a.x); f[base + 1] = cvt_bf16(a.y);
  f[base + 2] = cvt_bf16(a.z); f[base + 3] = cvt_bf16(a.w);
  f[base + 4] = cvt_bf16(b.x); f[base + 5] = cvt_bf16(b.y);
  f[base + 6] = cvt_bf16(b.z); f[base + 7] = cvt_bf16(b.w);
}

// ---------------- constants ----------------
static constexpr int Dm   = 256;
static constexpr int NHh  = 8;
static constexpr int HDh  = 32;
static constexpr int NCc  = 80;
static constexpr int BSz  = 7;    // box pool S
static constexpr int MSz  = 14;   // mask pool S
static constexpr int FFf  = 2048;
static constexpr int DDk  = 64;
static constexpr int Bn   = 2;
static constexpr int Nn   = 100;
static constexpr int Mp   = Bn * Nn;      // 200 proposals
static constexpr int Hf   = 96, Wf = 96;  // feature map
static constexpr int STG  = 6;
#define SCALE_CLAMP 4.1351665567423560f   // log(1000/16)

// ======================================================================
// Generic register-blocked WMMA GEMM.
//   C[M,N] = A[M,K] * B + bias, bf16 inputs, f32 accumulate.
//   A row-major (lda multiple of 4, rows 16B-aligned).
//   B element (k,n) at B[k*bRS + n*bCS].
//   Wave computes (MT*16) x (NT*16); out-of-range A rows are clamped
//   (their garbage never stored) -> zero branches in the hot loop.
//   outMode 0: row-major | 1: CHW scatter | 2: up-conv scatter
// K multiple of 32; N multiple of 16*NT.
// ======================================================================
template <int MT, int NT, bool BVEC>
__global__ void wmma_gemm_kernel(
    const float* __restrict__ A, long long aBS, int lda,
    const float* __restrict__ B, long long bBS, int bRS, int bCS,
    const float* __restrict__ bias,
    float* __restrict__ C, long long cBS,
    int Mrows, int Ncols, int K,
    int outMode, int relu, int npix, int ua, int ub)
{
  const float* Ab = A + (size_t)blockIdx.z * (size_t)aBS;
  const float* Bb = B + (size_t)blockIdx.z * (size_t)bBS;
  float*       Cb = C + (size_t)blockIdx.z * (size_t)cBS;

  const int lane = threadIdx.x & 31;
  const int hl   = lane >> 4;
  const int l16  = lane & 15;
  const int tM   = blockIdx.y * (MT * 16);
  const int tN   = blockIdx.x * (NT * 16);

  int arow[MT];
#pragma unroll
  for (int i = 0; i < MT; ++i) {
    int r = tM + i * 16 + l16;
    arow[i] = (r < Mrows) ? r : (Mrows - 1);
  }

  f32x8 acc[MT][NT];
  {
    f32x8 z = {};
#pragma unroll
    for (int i = 0; i < MT; ++i)
#pragma unroll
      for (int j = 0; j < NT; ++j) acc[i][j] = z;
  }

  for (int kb = 0; kb < K; kb += 32) {
    if (kb + 128 < K)
      __builtin_prefetch(Ab + (size_t)arow[0] * lda + kb + 128);

    bf16x16 af[MT];
#pragma unroll
    for (int i = 0; i < MT; ++i) {
      const float* rp = Ab + (size_t)arow[i] * lda + kb + hl * 16;
      const float4* p0 = (const float4*)rp;
      const float4* p1 = (const float4*)(rp + 8);
      float4 a0 = p0[0], a1 = p0[1];
      float4 b0 = p1[0], b1 = p1[1];
      pack8(af[i], 0, a0, a1);
      pack8(af[i], 8, b0, b1);
    }

#pragma unroll
    for (int j = 0; j < NT; ++j) {
      bf16x16 bf;
      const int col = tN + j * 16 + l16;
      if (BVEC) {
        const float* bp = Bb + (size_t)(kb + hl * 16) + (size_t)col * bCS;
        const float4* q = (const float4*)bp;
        float4 b0 = q[0], b1 = q[1], b2 = q[2], b3 = q[3];
        pack8(bf, 0, b0, b1);
        pack8(bf, 8, b2, b3);
      } else {
#pragma unroll
        for (int e = 0; e < 16; ++e) {
          int kB = kb + e + (hl << 4);
          bf[e] = cvt_bf16(Bb[(size_t)kB * bRS + (size_t)col * bCS]);
        }
      }
#pragma unroll
      for (int i = 0; i < MT; ++i)
        acc[i][j] = __builtin_amdgcn_wmma_f32_16x16x32_bf16(
            false, af[i], false, bf, (short)0, acc[i][j], false, false);
    }
  }

#pragma unroll
  for (int i = 0; i < MT; ++i)
#pragma unroll
    for (int j = 0; j < NT; ++j)
#pragma unroll
      for (int vi = 0; vi < 8; ++vi) {
        int row = tM + i * 16 + vi + (hl << 3);
        int col = tN + j * 16 + l16;
        if (row < Mrows) {
          float val = acc[i][j][vi];
          if (bias) val += bias[col];
          if (relu) val = fmaxf(val, 0.0f);
          size_t idx;
          if (outMode == 0) {
            idx = (size_t)row * Ncols + col;
          } else if (outMode == 1) {
            int m = row / npix, pix = row - m * npix;
            idx = ((size_t)m * Ncols + col) * (size_t)npix + pix;
          } else {
            int m = row / 196, pix = row - m * 196;
            int ii = pix / 14, jj = pix - ii * 14;
            idx = ((size_t)m * 784 + (size_t)(2 * ii + ua) * 28 + (2 * jj + ub)) * (size_t)Ncols + col;
          }
          Cb[idx] = val;
        }
      }
}

// ======================================================================
// conv3x3 weight repack: OIHW cw[o][c][ky][kx] -> k-major w2[(pos*256+c)*256+o]
// so the conv's implicit-GEMM K order is pos-major (tap-uniform per K step).
// ======================================================================
__global__ void conv_w_repack_kernel(const float* __restrict__ cw, float* __restrict__ w2)
{
  int tid = blockIdx.x * blockDim.x + threadIdx.x;
  if (tid >= 2304 * 256) return;
  int o  = tid & 255;
  int kk = tid >> 8;            // kk = pos*256 + c
  int pos = kk >> 8;
  int c   = kk & 255;
  w2[tid] = cw[(size_t)o * 2304 + c * 9 + pos];
}

// ======================================================================
// conv3x3 SAME as implicit GEMM with WMMA, pos-major K (tap uniform per
// 32-wide K step -> A gather is contiguous b128 runs from a shifted base,
// one boolean zero-pad select per fragment). 64x32 register blocking.
// in/out pixel-major (M, 196, 256); w2 k-major (2304, 256).
// ======================================================================
__global__ void conv3x3_wmma_kernel(
    const float* __restrict__ in, const float* __restrict__ w2,
    const float* __restrict__ bias, float* __restrict__ out, int Mtot)
{
  constexpr int MT = 4, NT = 2;
  const int lane = threadIdx.x & 31;
  const int hl   = lane >> 4;
  const int l16  = lane & 15;
  const int tM   = blockIdx.y * (MT * 16);
  const int tN   = blockIdx.x * (NT * 16);

  int mI[MT], yI[MT], xI[MT];
#pragma unroll
  for (int i = 0; i < MT; ++i) {
    int r = tM + i * 16 + l16;
    int rc = (r < Mtot) ? r : (Mtot - 1);
    mI[i] = rc / 196;
    int pix = rc - mI[i] * 196;
    yI[i] = pix / 14;
    xI[i] = pix - yI[i] * 14;
  }

  f32x8 acc[MT][NT];
  {
    f32x8 z = {};
#pragma unroll
    for (int i = 0; i < MT; ++i)
#pragma unroll
      for (int j = 0; j < NT; ++j) acc[i][j] = z;
  }

  for (int pos = 0; pos < 9; ++pos) {
    const int dy = pos / 3 - 1;
    const int dx = pos - (pos / 3) * 3 - 1;
    const float* abase[MT];
    bool valid[MT];
#pragma unroll
    for (int i = 0; i < MT; ++i) {
      int iy = yI[i] + dy, ix = xI[i] + dx;
      valid[i] = (iy >= 0) && (iy < 14) && (ix >= 0) && (ix < 14);
      int iyc = min(max(iy, 0), 13), ixc = min(max(ix, 0), 13);
      abase[i] = in + ((size_t)mI[i] * 196 + iyc * 14 + ixc) * 256;
    }
    for (int cb = 0; cb < 256; cb += 32) {
      bf16x16 af[MT];
      const bf16x16 zf = {};
#pragma unroll
      for (int i = 0; i < MT; ++i) {
        const float* rp = abase[i] + cb + hl * 16;
        const float4* p0 = (const float4*)rp;
        const float4* p1 = (const float4*)(rp + 8);
        float4 a0 = p0[0], a1 = p0[1];
        float4 b0 = p1[0], b1 = p1[1];
        pack8(af[i], 0, a0, a1);
        pack8(af[i], 8, b0, b1);
        af[i] = valid[i] ? af[i] : zf;
      }
      const int kb = pos * 256 + cb;
#pragma unroll
      for (int j = 0; j < NT; ++j) {
        const int col = tN + j * 16 + l16;
        bf16x16 bf;
#pragma unroll
        for (int e = 0; e < 16; ++e) {
          int kB = kb + e + (hl << 4);
          bf[e] = cvt_bf16(w2[(size_t)kB * 256 + col]);
        }
#pragma unroll
        for (int i = 0; i < MT; ++i)
          acc[i][j] = __builtin_amdgcn_wmma_f32_16x16x32_bf16(
              false, af[i], false, bf, (short)0, acc[i][j], false, false);
      }
    }
  }

#pragma unroll
  for (int i = 0; i < MT; ++i)
#pragma unroll
    for (int j = 0; j < NT; ++j)
#pragma unroll
      for (int vi = 0; vi < 8; ++vi) {
        int r = tM + i * 16 + vi + (hl << 3);
        int col = tN + j * 16 + l16;
        if (r < Mtot) out[(size_t)r * 256 + col] = acc[i][j][vi] + bias[col];
      }
}

// ---------------- LayerNorm (one wave per row), optional residual/relu ----
__global__ void ln_kernel(const float* __restrict__ x, const float* __restrict__ res,
                          const float* __restrict__ w, const float* __restrict__ b,
                          float* __restrict__ y, int cols, int relu)
{
  size_t row = blockIdx.x;
  const float* xr = x + row * cols;
  const float* rr = res ? res + row * cols : nullptr;
  int lane = threadIdx.x;
  float s = 0.f, s2 = 0.f;
  for (int c = lane; c < cols; c += 32) {
    float v = xr[c] + (rr ? rr[c] : 0.f);
    s += v; s2 += v * v;
  }
  for (int o = 16; o > 0; o >>= 1) { s += __shfl_xor(s, o); s2 += __shfl_xor(s2, o); }
  float mean = s / cols;
  float var  = s2 / cols - mean * mean;
  float rs   = rsqrtf(var + 1e-5f);
  for (int c = lane; c < cols; c += 32) {
    float v = xr[c] + (rr ? rr[c] : 0.f);
    float o_ = (v - mean) * rs * w[c] + b[c];
    if (relu) o_ = fmaxf(o_, 0.f);
    y[row * cols + c] = o_;
  }
}

// ---------------- features NCHW -> NHWC (coalesced ROI gathers) ----------
__global__ void feat_to_hwc_kernel(const float* __restrict__ in, float* __restrict__ out)
{
  size_t tid = (size_t)blockIdx.x * blockDim.x + threadIdx.x;
  size_t total = (size_t)Bn * Dm * Hf * Wf;
  if (tid >= total) return;
  int x = tid % Wf; size_t t = tid / Wf;
  int y = t % Hf;   t /= Hf;
  int c = t % Dm;   int b = t / Dm;
  out[(((size_t)b * Hf + y) * Wf + x) * Dm + c] = in[tid];
}

// ---------------- ROI align (RATIO=2 avg), pixel-major output ------------
__global__ void roi_align_kernel(const float* __restrict__ feat_hwc,
                                 const float* __restrict__ boxes,
                                 float* __restrict__ out, int S, size_t total)
{
  size_t tid = (size_t)blockIdx.x * blockDim.x + threadIdx.x;
  if (tid >= total) return;
  int c = tid & 255;
  size_t t = tid >> 8;
  int ss = t % (S * S);
  int m  = t / (S * S);
  int b  = m / Nn;
  const float* bx = boxes + (size_t)m * 4;
  float x1 = bx[0] * 0.125f - 0.5f, y1 = bx[1] * 0.125f - 0.5f;
  float x2 = bx[2] * 0.125f - 0.5f, y2 = bx[3] * 0.125f - 0.5f;
  float bw = (x2 - x1) / S, bh = (y2 - y1) / S;
  int py = ss / S, px = ss - py * S;
  const float* fb = feat_hwc + (size_t)b * Hf * Wf * Dm;
  float accv = 0.f;
#pragma unroll
  for (int ry = 0; ry < 2; ++ry) {
#pragma unroll
    for (int rx = 0; rx < 2; ++rx) {
      float offy = py + (ry + 0.5f) * 0.5f;
      float offx = px + (rx + 0.5f) * 0.5f;
      float ys = y1 + offy * bh, xs = x1 + offx * bw;
      bool valid = (ys >= -1.f) && (ys <= (float)Hf) && (xs >= -1.f) && (xs <= (float)Wf);
      float yc = fminf(fmaxf(ys, 0.f), (float)(Hf - 1));
      float xc = fminf(fmaxf(xs, 0.f), (float)(Wf - 1));
      int y0 = (int)floorf(yc), x0 = (int)floorf(xc);
      int y1i = min(y0 + 1, Hf - 1), x1i = min(x0 + 1, Wf - 1);
      float ly = yc - y0, lx = xc - x0, hy = 1.f - ly, hx = 1.f - lx;
      float v = fb[((size_t)y0 * Wf + x0) * Dm + c] * hy * hx
              + fb[((size_t)y0 * Wf + x1i) * Dm + c] * hy * lx
              + fb[((size_t)y1i * Wf + x0) * Dm + c] * ly * hx
              + fb[((size_t)y1i * Wf + x1i) * Dm + c] * ly * lx;
      accv += valid ? v : 0.f;
    }
  }
  out[tid] = accv * 0.25f;
}

// ---------------- pro init: mean over 49 spatial ------------------------
__global__ void mean_spatial_kernel(const float* __restrict__ rb, float* __restrict__ pro)
{
  int tid = blockIdx.x * blockDim.x + threadIdx.x;
  if (tid >= Mp * Dm) return;
  int m = tid >> 8, c = tid & 255;
  float s = 0.f;
  for (int p = 0; p < BSz * BSz; ++p) s += rb[((size_t)m * 49 + p) * Dm + c];
  pro[tid] = s * (1.0f / 49.0f);
}

// ---------------- fused MHA attention: per (q,h,b) block -----------------
__global__ void attn_kernel(const float* __restrict__ q, const float* __restrict__ k,
                            const float* __restrict__ v, float* __restrict__ out)
{
  __shared__ float sc[Nn];
  __shared__ float invSum;
  int qi = blockIdx.x, h = blockIdx.y, b = blockIdx.z;
  int m = b * Nn + qi;
  int t = threadIdx.x;
  const float* qp = q + (size_t)m * Dm + h * HDh;
  if (t < Nn) {
    const float* kp = k + (size_t)(b * Nn + t) * Dm + h * HDh;
    float s = 0.f;
    for (int d = 0; d < HDh; ++d) s += qp[d] * kp[d];
    sc[t] = s * 0.17677669529663687f;  // 1/sqrt(32)
  }
  __syncthreads();
  if (t == 0) {
    float mx = -1e30f;
    for (int i = 0; i < Nn; ++i) mx = fmaxf(mx, sc[i]);
    float sm = 0.f;
    for (int i = 0; i < Nn; ++i) { float e = __expf(sc[i] - mx); sc[i] = e; sm += e; }
    invSum = 1.f / sm;
  }
  __syncthreads();
  if (t < HDh) {
    float inv = invSum, accv = 0.f;
    for (int i = 0; i < Nn; ++i)
      accv += sc[i] * v[(size_t)(b * Nn + i) * Dm + h * HDh + t];
    out[(size_t)m * Dm + h * HDh + t] = accv * inv;
  }
}

// ---------------- time embedding MLP: ss = silu(te) @ t_w + t_b ----------
__global__ void time_mlp_kernel(const float* __restrict__ te, const float* __restrict__ tw,
                                const float* __restrict__ tb, float* __restrict__ ss)
{
  int tid = blockIdx.x * blockDim.x + threadIdx.x;
  if (tid >= Bn * 2 * Dm) return;
  int b = tid >> 9, j = tid & 511;
  const float* tr = te + (size_t)b * 4 * Dm;
  float accv = tb[j];
  for (int i = 0; i < 4 * Dm; ++i) {
    float xx = tr[i];
    accv += (xx / (1.f + __expf(-xx))) * tw[(size_t)i * (2 * Dm) + j];
  }
  ss[tid] = accv;
}

// ---------------- fc = obj*(scale+1)+shift -------------------------------
__global__ void scale_shift_kernel(const float* __restrict__ obj, const float* __restrict__ ss,
                                   float* __restrict__ out)
{
  int tid = blockIdx.x * blockDim.x + threadIdx.x;
  if (tid >= Mp * Dm) return;
  int m = tid >> 8, c = tid & 255;
  int b = m / Nn;
  out[tid] = obj[tid] * (ss[(size_t)b * 512 + c] + 1.f) + ss[(size_t)b * 512 + 256 + c];
}

// ---------------- reg head (N=4 too small for WMMA) ----------------------
__global__ void reg_out_kernel(const float* __restrict__ rf, const float* __restrict__ rw,
                               const float* __restrict__ rb_, float* __restrict__ deltas)
{
  int tid = blockIdx.x * blockDim.x + threadIdx.x;
  if (tid >= Mp * 4) return;
  int m = tid >> 2, j = tid & 3;
  float accv = rb_[j];
  for (int d = 0; d < Dm; ++d) accv += rf[(size_t)m * Dm + d] * rw[(size_t)d * 4 + j];
  deltas[tid] = accv;
}

// ---------------- apply deltas, write pb to d_out & update boxes ---------
__global__ void apply_deltas_kernel(const float* __restrict__ deltas,
                                    float* __restrict__ boxes_cur, float* __restrict__ pb_out)
{
  int m = blockIdx.x * blockDim.x + threadIdx.x;
  if (m >= Mp) return;
  float b0 = boxes_cur[m * 4 + 0], b1 = boxes_cur[m * 4 + 1];
  float b2 = boxes_cur[m * 4 + 2], b3 = boxes_cur[m * 4 + 3];
  float w = b2 - b0, h = b3 - b1;
  float cx = b0 + 0.5f * w, cy = b1 + 0.5f * h;
  float dx = deltas[m * 4 + 0] * 0.5f, dy = deltas[m * 4 + 1] * 0.5f;
  float dw = fminf(deltas[m * 4 + 2], SCALE_CLAMP);
  float dh = fminf(deltas[m * 4 + 3], SCALE_CLAMP);
  float pcx = dx * w + cx, pcy = dy * h + cy;
  float pw = __expf(dw) * w, ph = __expf(dh) * h;
  float o0 = pcx - 0.5f * pw, o1 = pcy - 0.5f * ph;
  float o2 = pcx + 0.5f * pw, o3 = pcy + 0.5f * ph;
  pb_out[m * 4 + 0] = o0; pb_out[m * 4 + 1] = o1;
  pb_out[m * 4 + 2] = o2; pb_out[m * 4 + 3] = o3;
  boxes_cur[m * 4 + 0] = o0; boxes_cur[m * 4 + 1] = o1;
  boxes_cur[m * 4 + 2] = o2; boxes_cur[m * 4 + 3] = o3;
}

__global__ void copy_kernel(float* __restrict__ dst, const float* __restrict__ src, int n)
{
  int i = blockIdx.x * blockDim.x + threadIdx.x;
  if (i < n) dst[i] = src[i];
}

// ======================================================================
// Host side
// ======================================================================
static inline void gemm(hipStream_t st, const float* A, long long aBS, int lda,
                        const float* B, long long bBS, int bRS, int bCS,
                        const float* bias, float* C, long long cBS,
                        int M, int N, int K, int batch,
                        int outMode = 0, int relu = 0, int npix = 1, int ua = 0, int ub = 0)
{
  if (N % 32 == 0) {
    dim3 g(N / 32, (M + 63) / 64, batch);
    wmma_gemm_kernel<4, 2, false><<<g, dim3(32), 0, st>>>(
        A, aBS, lda, B, bBS, bRS, bCS, bias, C, cBS, M, N, K, outMode, relu, npix, ua, ub);
  } else if (bRS == 1) {
    dim3 g(N / 16, (M + 63) / 64, batch);
    wmma_gemm_kernel<4, 1, true><<<g, dim3(32), 0, st>>>(
        A, aBS, lda, B, bBS, bRS, bCS, bias, C, cBS, M, N, K, outMode, relu, npix, ua, ub);
  } else {
    dim3 g(N / 16, (M + 63) / 64, batch);
    wmma_gemm_kernel<4, 1, false><<<g, dim3(32), 0, st>>>(
        A, aBS, lda, B, bBS, bRS, bCS, bias, C, cBS, M, N, K, outMode, relu, npix, ua, ub);
  }
}

static inline void ln(hipStream_t st, const float* x, const float* res,
                      const float* w, const float* b, float* y, int rows, int cols, int relu)
{
  ln_kernel<<<rows, 32, 0, st>>>(x, res, w, b, y, cols, relu);
}

struct BoxParams {
  const float *an_b, *an_w, *bk, *bo, *bq, *bv, *wk, *wo, *wq, *wv;
  const float *c_w, *cls_b, *cls_w, *cn_b, *cn_w;
  const float *dyn_b, *dyn_w, *fc_b, *fc_w, *n1_b, *n1_w, *n2_b, *n2_w, *n3_b, *n3_w;
  const float *f1_b, *f1_w, *f2_b, *f2_w, *fn_b, *fn_w, *in_b, *in_w;
  const float *rw[3], *rnw[3], *rnb[3];
  const float *reg_b, *reg_w, *t_b, *t_w;
};
struct MaskParams {
  const float *cw[4], *cb[4];
  const float *dyn_b, *dyn_w, *n1_b, *n1_w, *n2_b, *n2_w;
  const float *lg_b, *lg_w, *up_b, *up_w;
};

extern "C" void kernel_launch(void* const* d_in, const int* in_sizes, int n_in,
                              void* d_out, int out_size, void* d_ws, size_t ws_size,
                              hipStream_t stream)
{
  (void)in_sizes; (void)n_in; (void)out_size; (void)ws_size;
  auto F = [&](int i) -> const float* { return (const float*)d_in[i]; };

  const float* features = F(0);   // (2,256,96,96)
  const float* boxes_in = F(1);   // (2,100,4)
  const float* time_emb = F(2);   // (2,1024)

  // ---- param leaf map (JAX pytree: dicts in sorted-key order, lists in order)
  BoxParams BP[STG]; MaskParams MK[STG];
  for (int s = 0; s < STG; ++s) {
    int p = 3 + s * 46;
    BoxParams& q = BP[s];
    q.an_b = F(p + 0);  q.an_w = F(p + 1);
    q.bk = F(p + 2); q.bo = F(p + 3); q.bq = F(p + 4); q.bv = F(p + 5);
    q.wk = F(p + 6); q.wo = F(p + 7); q.wq = F(p + 8); q.wv = F(p + 9);
    q.c_w = F(p + 10); q.cls_b = F(p + 11); q.cls_w = F(p + 12);
    q.cn_b = F(p + 13); q.cn_w = F(p + 14);
    q.dyn_b = F(p + 15); q.dyn_w = F(p + 16); q.fc_b = F(p + 17); q.fc_w = F(p + 18);
    q.n1_b = F(p + 19); q.n1_w = F(p + 20); q.n2_b = F(p + 21); q.n2_w = F(p + 22);
    q.n3_b = F(p + 23); q.n3_w = F(p + 24);
    q.f1_b = F(p + 25); q.f1_w = F(p + 26); q.f2_b = F(p + 27); q.f2_w = F(p + 28);
    q.fn_b = F(p + 29); q.fn_w = F(p + 30); q.in_b = F(p + 31); q.in_w = F(p + 32);
    for (int r = 0; r < 3; ++r) {
      q.rw[r] = F(p + 33 + r * 3); q.rnw[r] = F(p + 34 + r * 3); q.rnb[r] = F(p + 35 + r * 3);
    }
    q.reg_b = F(p + 42); q.reg_w = F(p + 43); q.t_b = F(p + 44); q.t_w = F(p + 45);
  }
  for (int s = 0; s < STG; ++s) {
    int p = 3 + STG * 46 + s * 18;
    MaskParams& q = MK[s];
    for (int c = 0; c < 4; ++c) { q.cw[c] = F(p + c * 2); q.cb[c] = F(p + c * 2 + 1); }
    q.dyn_b = F(p + 8); q.dyn_w = F(p + 9);
    q.n1_b = F(p + 10); q.n1_w = F(p + 11); q.n2_b = F(p + 12); q.n2_w = F(p + 13);
    q.lg_b = F(p + 14); q.lg_w = F(p + 15); q.up_b = F(p + 16); q.up_w = F(p + 17);
  }

  // ---- d_out layout: logits (6*16000) | pboxes (6*800) | masks (6*12,544,000)
  float* out = (float*)d_out;
  float* out_logits = out;
  float* out_pbox   = out + STG * Mp * NCc;            // 96000
  float* out_masks  = out + STG * Mp * NCc + STG * Mp * 4;

  // ---- workspace bump allocator (floats)
  float* wsf = (float*)d_ws;
  size_t pos = 0;
  auto alloc = [&](size_t n) { float* p = wsf + pos; pos += (n + 63) & ~(size_t)63; return p; };
  float* boxes_cur = alloc(Mp * 4);
  float* feat_hwc  = alloc((size_t)Bn * Dm * Hf * Wf);
  float* rb   = alloc((size_t)Mp * 49 * Dm);
  float* rm   = alloc((size_t)Mp * 196 * Dm);
  float* pro  = alloc((size_t)Mp * Dm);
  float* x1   = alloc((size_t)Mp * Dm);
  float* qb   = alloc((size_t)Mp * Dm);
  float* kb   = alloc((size_t)Mp * Dm);
  float* vb   = alloc((size_t)Mp * Dm);
  float* att  = alloc((size_t)Mp * Dm);
  float* tmp1 = alloc((size_t)Mp * Dm);
  float* tmp2 = alloc((size_t)Mp * Dm);
  float* prm  = alloc((size_t)Mp * 2 * Dm * DDk);   // 200 x 32768
  float* f1b  = alloc((size_t)Mp * 196 * DDk);
  float* f2b  = alloc((size_t)Mp * 196 * Dm);       // conv ping
  float* xb1  = alloc((size_t)Mp * 196 * Dm);       // conv pong
  float* iic  = alloc((size_t)Mp * Dm);
  float* ffnh = alloc((size_t)Mp * FFf);
  float* ssb  = alloc((size_t)Bn * 2 * Dm);
  float* fcf  = alloc((size_t)Mp * Dm);
  float* delt = alloc((size_t)Mp * 4);
  float* up   = alloc((size_t)Mp * 784 * Dm);
  float* w2   = alloc((size_t)2304 * 256);          // repacked conv weights

  // ---- one-time prep
  copy_kernel<<<(Mp * 4 + 255) / 256, 256, 0, stream>>>(boxes_cur, boxes_in, Mp * 4);
  {
    size_t tot = (size_t)Bn * Dm * Hf * Wf;
    feat_to_hwc_kernel<<<(tot + 255) / 256, 256, 0, stream>>>(features, feat_hwc);
  }

  for (int s = 0; s < STG; ++s) {
    const BoxParams& P = BP[s];
    const MaskParams& Q = MK[s];

    // ---- ROI pools (pixel-major (M,S*S,256)) with current boxes
    {
      size_t tb = (size_t)Mp * 49 * Dm;
      roi_align_kernel<<<(tb + 255) / 256, 256, 0, stream>>>(feat_hwc, boxes_cur, rb, BSz, tb);
      size_t tm = (size_t)Mp * 196 * Dm;
      roi_align_kernel<<<(tm + 255) / 256, 256, 0, stream>>>(feat_hwc, boxes_cur, rm, MSz, tm);
    }
    if (s == 0)
      mean_spatial_kernel<<<(Mp * Dm + 255) / 256, 256, 0, stream>>>(rb, pro);

    // ---- MHA: q,k,v = pro@W + b ; fused softmax-attn ; out proj ; LN(an)
    gemm(stream, pro, 0, Dm, P.wq, 0, Dm, 1, P.bq, qb, 0, Mp, Dm, Dm, 1);
    gemm(stream, pro, 0, Dm, P.wk, 0, Dm, 1, P.bk, kb, 0, Mp, Dm, Dm, 1);
    gemm(stream, pro, 0, Dm, P.wv, 0, Dm, 1, P.bv, vb, 0, Mp, Dm, Dm, 1);
    attn_kernel<<<dim3(Nn, NHh, Bn), 128, 0, stream>>>(qb, kb, vb, att);
    gemm(stream, att, 0, Dm, P.wo, 0, Dm, 1, P.bo, tmp1, 0, Mp, Dm, Dm, 1);
    ln(stream, tmp1, pro, P.an_w, P.an_b, x1, Mp, Dm, 0);

    // ---- box dynamic conv (with projection)
    gemm(stream, x1, 0, Dm, P.dyn_w, 0, 2 * Dm * DDk, 1, P.dyn_b, prm, 0, Mp, 2 * Dm * DDk, Dm, 1);
    gemm(stream, rb, 49 * Dm, Dm, prm, 2 * Dm * DDk, DDk, 1, nullptr, f1b, 49 * DDk, 49, DDk, Dm, Mp);
    ln(stream, f1b, nullptr, P.n1_w, P.n1_b, f1b, Mp * 49, DDk, 1);
    gemm(stream, f1b, 49 * DDk, DDk, prm + Dm * DDk, 2 * Dm * DDk, Dm, 1, nullptr, f2b, 49 * Dm, 49, Dm, DDk, Mp);
    ln(stream, f2b, nullptr, P.n2_w, P.n2_b, f2b, Mp * 49, Dm, 1);
    gemm(stream, f2b, 0, 49 * Dm, P.fc_w, 0, Dm, 1, P.fc_b, iic, 0, Mp, Dm, 49 * Dm, 1);
    ln(stream, iic, nullptr, P.n3_w, P.n3_b, iic, Mp, Dm, 1);
    ln(stream, x1, iic, P.in_w, P.in_b, tmp2, Mp, Dm, 0);  // obj pre-FFN in tmp2

    // ---- FFN + LN(fn) -> new pro (the "obj" output)
    gemm(stream, tmp2, 0, Dm, P.f1_w, 0, FFf, 1, P.f1_b, ffnh, 0, Mp, FFf, Dm, 1, 0, 1);
    gemm(stream, ffnh, 0, FFf, P.f2_w, 0, Dm, 1, P.f2_b, tmp1, 0, Mp, Dm, FFf, 1);
    ln(stream, tmp2, tmp1, P.fn_w, P.fn_b, pro, Mp, Dm, 0);

    // ---- time conditioning
    time_mlp_kernel<<<(Bn * 2 * Dm + 127) / 128, 128, 0, stream>>>(time_emb, P.t_w, P.t_b, ssb);
    scale_shift_kernel<<<(Mp * Dm + 255) / 256, 256, 0, stream>>>(pro, ssb, fcf);

    // ---- cls tower -> logits (stage slab of d_out)
    gemm(stream, fcf, 0, Dm, P.c_w, 0, Dm, 1, nullptr, tmp1, 0, Mp, Dm, Dm, 1);
    ln(stream, tmp1, nullptr, P.cn_w, P.cn_b, tmp1, Mp, Dm, 1);
    gemm(stream, tmp1, 0, Dm, P.cls_w, 0, NCc, 1, P.cls_b,
         out_logits + (size_t)s * Mp * NCc, 0, Mp, NCc, Dm, 1);

    // ---- reg tower (3x [GEMM + LN relu]) -> deltas -> boxes
    const float* cur = fcf;
    float* pp[2] = {tmp1, tmp2};
    for (int r = 0; r < 3; ++r) {
      float* nxt = pp[r & 1];
      gemm(stream, cur, 0, Dm, P.rw[r], 0, Dm, 1, nullptr, nxt, 0, Mp, Dm, Dm, 1);
      ln(stream, nxt, nullptr, P.rnw[r], P.rnb[r], nxt, Mp, Dm, 1);
      cur = nxt;
    }
    reg_out_kernel<<<(Mp * 4 + 127) / 128, 128, 0, stream>>>(cur, P.reg_w, P.reg_b, delt);
    apply_deltas_kernel<<<(Mp + 63) / 64, 64, 0, stream>>>(delt, boxes_cur,
                                                           out_pbox + (size_t)s * Mp * 4);

    // ---- mask head: dynamic conv (no projection), pixel-major throughout
    gemm(stream, pro, 0, Dm, Q.dyn_w, 0, 2 * Dm * DDk, 1, Q.dyn_b, prm, 0, Mp, 2 * Dm * DDk, Dm, 1);
    gemm(stream, rm, 196 * Dm, Dm, prm, 2 * Dm * DDk, DDk, 1, nullptr, f1b, 196 * DDk, 196, DDk, Dm, Mp);
    ln(stream, f1b, nullptr, Q.n1_w, Q.n1_b, f1b, Mp * 196, DDk, 1);
    gemm(stream, f1b, 196 * DDk, DDk, prm + Dm * DDk, 2 * Dm * DDk, Dm, 1, nullptr, f2b, 196 * Dm, 196, Dm, DDk, Mp);
    ln(stream, f2b, nullptr, Q.n2_w, Q.n2_b, f2b, Mp * 196, Dm, 1);

    // ---- 4x conv3x3 (repack weights to k-major, then implicit GEMM)
    {
      const int Mtot = Mp * 196;
      dim3 gc(Dm / 32, (Mtot + 63) / 64);
      const int rpThreads = 2304 * 256;
      float* pin[2] = {f2b, xb1};
      for (int c = 0; c < 4; ++c) {
        conv_w_repack_kernel<<<(rpThreads + 255) / 256, 256, 0, stream>>>(Q.cw[c], w2);
        conv3x3_wmma_kernel<<<gc, dim3(32), 0, stream>>>(pin[c & 1], w2, Q.cb[c],
                                                         pin[(c + 1) & 1], Mtot);
      }
    }

    // ---- 2x2 stride-2 transposed conv as 4 GEMMs with scatter epilogue (+relu)
    for (int a = 0; a < 2; ++a)
      for (int bq2 = 0; bq2 < 2; ++bq2)
        gemm(stream, f2b, 0, Dm, Q.up_w + (a * 2 + bq2), 0, 4 * Dm, 4, Q.up_b,
             up, 0, Mp * 196, Dm, Dm, 1, /*outMode*/2, /*relu*/1, 196, a, bq2);

    // ---- 1x1 conv to NC=80 classes, CHW scatter straight into d_out masks
    gemm(stream, up, 0, Dm, Q.lg_w, 0, 1, Dm, Q.lg_b,
         out_masks + (size_t)s * Mp * NCc * 784, 0, Mp * 784, NCc, Dm, 1,
         /*outMode*/1, 0, /*npix*/784);
  }
}
